// MoELayer_17927193494290
// MI455X (gfx1250) — matrix operand order repaired and checked by reading, not compiled, not run on hardware
//
#include <hip/hip_runtime.h>
#include <hip/hip_bf16.h>

// ---------------------------------------------------------------------------
// MoE layer for MI455X (gfx1250): routed top-2, fused bf16-WMMA expert FFN,
// pre-transposed bf16 weights, async global->LDS B128 staging.
// ---------------------------------------------------------------------------

#define N_EXPERTS 8
#define S_DIM 2048
#define B_DIM 8
#define D_DIM 1024
#define F_DIM 4096
#define TOKENS (S_DIM * B_DIM)          // 16384
#define TILE_M 32                        // tokens per workgroup
#define FC 64                            // F-chunk per iteration
#define MAX_TILES (TOKENS / TILE_M)      // 512

typedef unsigned short u16;
typedef unsigned int   u32;
typedef __attribute__((ext_vector_type(16))) __bf16 bf16x16;
typedef __attribute__((ext_vector_type(8)))  float  floatx8;
typedef __attribute__((ext_vector_type(8)))  u32    uintx8;

// LDS strides (in u16 elements), padded to break bank conflicts (64 banks x 4B)
#define XS_STRIDE   (D_DIM + 8)          // 1032 : X tile row-major [row][k]
#define W1T_STRIDE  (D_DIM + 8)          // 1032 : W1 chunk stored [f-col][k]
#define W2T_STRIDE  (FC + 8)             // 72   : W2 chunk stored [d-col][k]
#define HC_STRIDE   (FC + 8)             // 72   : Hc stored row-major [row][f]

// LDS layout (bytes)
#define LDS_TOKS_OFF   0                                 // 32 * int
#define LDS_WTS_OFF    128                               // 32 * float
#define LDS_XS_OFF     256                               // 32*1032*2 = 66048
#define LDS_HC_OFF     (256 + TILE_M * XS_STRIDE * 2)    // 66304
#define LDS_WT_OFF     (LDS_HC_OFF + TILE_M * HC_STRIDE * 2)  // 70912
#define LDS_WT_BYTES   (D_DIM * W2T_STRIDE * 2)          // 147456 (max of W1t/W2t)
#define LDS_TOTAL      (LDS_WT_OFF + LDS_WT_BYTES)       // 218368 < 320KB WGP LDS

// ---------------------------------------------------------------------------
// Async global->LDS (gfx1250): B128 per lane, tracked with ASYNCcnt.
// Builtin signature (from probe diagnostic): first param is
// 'int __vector(4) AS(1)*' (global src), second is the AS(3) LDS dst.
// Fallback: plain uint4 copy through VGPRs.
// ---------------------------------------------------------------------------
#if __has_builtin(__builtin_amdgcn_global_load_async_to_lds_b128)
#define USE_ASYNC_LDS 1
#endif

typedef int i32x4v __attribute__((__vector_size__(16)));

__device__ __forceinline__ void cp_b128(void* lds_dst, const void* gsrc) {
#if defined(USE_ASYNC_LDS)
  __builtin_amdgcn_global_load_async_to_lds_b128(
      (__attribute__((address_space(1))) i32x4v*)(uintptr_t)gsrc,
      (__attribute__((address_space(3))) i32x4v*)(unsigned int)(uintptr_t)lds_dst,
      0, 0);
#else
  *(uint4*)lds_dst = *(const uint4*)gsrc;
#endif
}

__device__ __forceinline__ void wait_async_lds() {
#if defined(USE_ASYNC_LDS)
  asm volatile("s_wait_asynccnt 0x0" ::: "memory");
#endif
}

__device__ __forceinline__ u16 f2bf(float f) {
  u32 x = __float_as_uint(f);
  u32 r = x + 0x7fffu + ((x >> 16) & 1u);   // round-to-nearest-even
  return (u16)(r >> 16);
}

__device__ __forceinline__ floatx8 zero8() {
  floatx8 z;
#pragma unroll
  for (int i = 0; i < 8; ++i) z[i] = 0.0f;
  return z;
}

__device__ __forceinline__ floatx8 wmma_bf16(uintx8 a, uintx8 b, floatx8 c) {
  return __builtin_amdgcn_wmma_f32_16x16x32_bf16(
      false, __builtin_bit_cast(bf16x16, a),
      false, __builtin_bit_cast(bf16x16, b),
      (short)0, c, false, false);
}

// A fragment (16x32 bf16, M x K), source row-major [row][k] in LDS.
// ISA 7.12.2: lane<16 holds K {0..7,16..23}; lane>=16 holds K {8..15,24..31}.
__device__ __forceinline__ uintx8 load_a_frag(const u32* lds, int row_base,
                                              int k0, int lane, int stride_us) {
  int row = row_base + (lane & 15);
  int kb  = k0 + ((lane & 16) ? 8 : 0);
  int base = row * stride_us;
  uintx8 r;
#pragma unroll
  for (int v = 0; v < 4; ++v) r[v]     = lds[(base + kb + 2 * v) >> 1];
#pragma unroll
  for (int v = 0; v < 4; ++v) r[4 + v] = lds[(base + kb + 16 + 2 * v) >> 1];
  return r;
}

// B fragment (32x16 bf16, K x N), source stored transposed [col][k] in LDS.
// lane<16 holds K {0..7,16..23}; lane>=16 holds K {8..15,24..31} (matches A).
__device__ __forceinline__ uintx8 load_b_frag(const u32* lds, int col_base,
                                              int k0, int lane, int stride_us) {
  int col = col_base + (lane & 15);
  int kb  = k0 + ((lane & 16) ? 8 : 0);
  int base = col * stride_us;
  uintx8 r;
#pragma unroll
  for (int v = 0; v < 8; ++v) r[v] = lds[(base + kb + 2 * (v & 3) + ((v & 4) ? 16 : 0)) >> 1];
  return r;
}

// ---------------------------------------------------------------------------
// Kernel 0: zero output + expert counters
// ---------------------------------------------------------------------------
__global__ void moe_zero_kernel(float* __restrict__ out, int n, int* __restrict__ cnt) {
  int gid = blockIdx.x * blockDim.x + threadIdx.x;
  if (gid < N_EXPERTS) cnt[gid] = 0;
  for (int i = gid; i < n; i += gridDim.x * blockDim.x) out[i] = 0.0f;
}

// ---------------------------------------------------------------------------
// Kernel 1a: f32 -> bf16 linear conversion (x)
// ---------------------------------------------------------------------------
__global__ void moe_cvt_kernel(const float* __restrict__ src, u16* __restrict__ dst, int n) {
  int gid = blockIdx.x * blockDim.x + threadIdx.x;
  for (int i = gid; i < n; i += gridDim.x * blockDim.x) dst[i] = f2bf(src[i]);
}

// ---------------------------------------------------------------------------
// Kernel 1b: f32 -> bf16 + transpose per expert. src[e]: RxC f32 row-major,
// dst[e]: CxR bf16 row-major. 64x64 tiles via LDS, coalesced on both sides.
// grid = ((R/64)*(C/64), N_EXPERTS), 256 threads.
// ---------------------------------------------------------------------------
__global__ __launch_bounds__(256)
void moe_cvt_tr_kernel(const float* __restrict__ src, u16* __restrict__ dst,
                       int R, int C) {
  __shared__ u16 T[64][72];               // padded: no store bank conflicts
  int e = blockIdx.y;
  int tilesC = C >> 6;
  int tr = (blockIdx.x / tilesC) << 6;    // source row-tile origin
  int tc = (blockIdx.x % tilesC) << 6;    // source col-tile origin
  const float* s = src + (size_t)e * R * C;
  u16* d = dst + (size_t)e * R * C;
  int tid = threadIdx.x;
#pragma unroll
  for (int it = 0; it < 16; ++it) {
    int i = tid + it * 256;
    int c = i & 63, r = i >> 6;           // coalesced f32 reads
    T[r][c] = f2bf(s[(size_t)(tr + r) * C + tc + c]);
  }
  __syncthreads();
#pragma unroll
  for (int it = 0; it < 16; ++it) {
    int i = tid + it * 256;
    int r = i & 63, c = i >> 6;           // coalesced u16 writes
    d[(size_t)(tc + c) * R + tr + r] = T[r][c];
  }
}

// ---------------------------------------------------------------------------
// Kernel 2: gating. One wave32 per token: 8 dots of length D, top-2 softmax,
// atomic-append into per-expert token lists.
// ---------------------------------------------------------------------------
__global__ __launch_bounds__(256)
void moe_gate_kernel(const float* __restrict__ x, const float* __restrict__ gw,
                     const float* __restrict__ gb, int* __restrict__ cnt,
                     int* __restrict__ lists, float* __restrict__ wlist) {
  int tid = threadIdx.x;
  int lane = tid & 31;
  int wv = tid >> 5;
  int token = blockIdx.x * 8 + wv;
  if (token >= TOKENS) return;

  float acc[N_EXPERTS];
#pragma unroll
  for (int e = 0; e < N_EXPERTS; ++e) acc[e] = 0.0f;

  const float* xr = x + (size_t)token * D_DIM;
  for (int d = lane; d < D_DIM; d += 32) {
    float xv = xr[d];
#pragma unroll
    for (int e = 0; e < N_EXPERTS; ++e) acc[e] += xv * gw[e * D_DIM + d];
  }
#pragma unroll
  for (int e = 0; e < N_EXPERTS; ++e) {
    float v = acc[e];
#pragma unroll
    for (int off = 16; off > 0; off >>= 1) v += __shfl_xor(v, off, 32);
    acc[e] = v + gb[e];
  }
  if (lane == 0) {
    int e0 = 0; float l0 = acc[0];
#pragma unroll
    for (int e = 1; e < N_EXPERTS; ++e)
      if (acc[e] > l0) { l0 = acc[e]; e0 = e; }
    int e1 = (e0 == 0) ? 1 : 0; float l1 = acc[e1];
#pragma unroll
    for (int e = 0; e < N_EXPERTS; ++e)
      if (e != e0 && acc[e] > l1) { l1 = acc[e]; e1 = e; }
    float ex = __expf(l1 - l0);           // l1 <= l0, stable
    float w0 = 1.0f / (1.0f + ex);
    float w1 = ex / (1.0f + ex);
    int p0 = atomicAdd(&cnt[e0], 1);
    lists[e0 * TOKENS + p0] = token;
    wlist[e0 * TOKENS + p0] = w0;
    int p1 = atomicAdd(&cnt[e1], 1);
    lists[e1 * TOKENS + p1] = token;
    wlist[e1 * TOKENS + p1] = w1;
  }
}

// ---------------------------------------------------------------------------
// Kernel 3: fused expert FFN. grid = (MAX_TILES, N_EXPERTS), 256 threads.
// w1t: [e][f][k] bf16 (pre-transposed), w2t: [e][d][f] bf16 (pre-transposed).
// Per workgroup: 32 gathered tokens; loop F in chunks of 64:
//   GEMM1: Hc = relu(Xs(32x1024) * W1c(1024x64) + b1)   [one 16x16 tile/wave]
//   GEMM2: Y(32x1024) += Hc(32x64) * W2c(64x1024)       [32x128 slab/wave]
// All chunk staging is contiguous B128 async global->LDS.
// Epilogue: out[tok] += w_tok * (Y + b2) via f32 global atomics.
// ---------------------------------------------------------------------------
__global__ __launch_bounds__(256)
void moe_ffn_kernel(const u16* __restrict__ xb, const u16* __restrict__ w1t,
                    const u16* __restrict__ w2t, const float* __restrict__ b1,
                    const float* __restrict__ b2, const int* __restrict__ cnt,
                    const int* __restrict__ lists, const float* __restrict__ wlist,
                    float* __restrict__ out) {
  extern __shared__ char smem[];
  int* toksL  = (int*)(smem + LDS_TOKS_OFF);
  float* wtsL = (float*)(smem + LDS_WTS_OFF);
  u16* Xs = (u16*)(smem + LDS_XS_OFF);
  u16* Hc = (u16*)(smem + LDS_HC_OFF);
  u16* Wt = (u16*)(smem + LDS_WT_OFF);
  const u32* XsU = (const u32*)Xs;
  const u32* HcU = (const u32*)Hc;
  const u32* WtU = (const u32*)Wt;

  int e = blockIdx.y;
  int tile = blockIdx.x;
  int count = cnt[e];
  if (tile * TILE_M >= count) return;   // uniform exit: EXEC all-1 for WMMA

  int tid = threadIdx.x;
  int lane = tid & 31;
  int wv = tid >> 5;

  // --- token list + gate weights for this tile (pad with weight 0) ---
  if (tid < TILE_M) {
    int idx = tile * TILE_M + tid;
    if (idx < count) {
      toksL[tid] = lists[e * TOKENS + idx];
      wtsL[tid]  = wlist[e * TOKENS + idx];
    } else {
      toksL[tid] = 0;
      wtsL[tid]  = 0.0f;
    }
  }
  __syncthreads();

  // --- gather X tile into LDS: 32 rows x 128 B128 copies (async) ---
  for (int i = tid; i < TILE_M * (D_DIM / 8); i += 256) {
    int r = i >> 7;
    int q = i & 127;
    int tok = toksL[r];
    cp_b128(&Xs[r * XS_STRIDE + 8 * q], &xb[(size_t)tok * D_DIM + 8 * q]);
  }

  // --- per-wave output accumulators: 32 rows x 128 cols (cols 128*wv..) ---
  floatx8 Y[2][8];
#pragma unroll
  for (int m = 0; m < 2; ++m)
#pragma unroll
    for (int n = 0; n < 8; ++n) Y[m][n] = zero8();

  int m1 = wv & 1;        // GEMM1 tile coords for this wave
  int n1 = wv >> 1;
  const size_t w1base = (size_t)e * F_DIM * D_DIM;   // [e][f][k]
  const size_t w2base = (size_t)e * D_DIM * F_DIM;   // [e][d][f]

  for (int fc = 0; fc < F_DIM; fc += FC) {
    __syncthreads();   // previous GEMM2 done reading Wt/Hc

    // W1 chunk: Wt[f-col][k] <- w1t[e][fc+j][0..1023], contiguous B128 rows
    for (int i = tid; i < FC * (D_DIM / 8); i += 256) {   // 8192 x 16B
      int j = i >> 7;
      int q = i & 127;
      cp_b128(&Wt[j * W1T_STRIDE + 8 * q],
              &w1t[w1base + (size_t)(fc + j) * D_DIM + 8 * q]);
    }
    wait_async_lds();   // covers Xs gather on first iteration too
    __syncthreads();

    // GEMM1: one 16x16 tile of Hc per wave, K = 1024
    floatx8 hacc = zero8();
    for (int k0 = 0; k0 < D_DIM; k0 += 32) {
      uintx8 af = load_a_frag(XsU, 16 * m1, k0, lane, XS_STRIDE);
      uintx8 bf = load_b_frag(WtU, 16 * n1, k0, lane, W1T_STRIDE);
      hacc = wmma_bf16(af, bf, hacc);
    }
    // bias + relu -> Hc (bf16)
    {
      int ncol = 16 * n1 + (lane & 15);
      float bb = b1[e * F_DIM + fc + ncol];
#pragma unroll
      for (int r = 0; r < 8; ++r) {
        float h = hacc[r] + bb;
        h = h > 0.0f ? h : 0.0f;
        int row = 16 * m1 + r + ((lane & 16) ? 8 : 0);
        Hc[row * HC_STRIDE + ncol] = f2bf(h);
      }
    }
    __syncthreads();   // Hc complete, GEMM1 done reading Wt

    // W2 chunk: Wt[d-col][k] <- w2t[e][col][fc..fc+63], contiguous B128 rows
    for (int i = tid; i < D_DIM * (FC / 8); i += 256) {   // 8192 x 16B
      int col = i >> 3;
      int q = i & 7;
      cp_b128(&Wt[col * W2T_STRIDE + 8 * q],
              &w2t[w2base + (size_t)col * F_DIM + fc + 8 * q]);
    }
    wait_async_lds();
    __syncthreads();

    // GEMM2: Y(32x128 slab) += Hc(32x64) * W2c(64x128 slab)
    for (int k0 = 0; k0 < FC; k0 += 32) {
      uintx8 a0 = load_a_frag(HcU, 0, k0, lane, HC_STRIDE);
      uintx8 a1 = load_a_frag(HcU, 16, k0, lane, HC_STRIDE);
#pragma unroll
      for (int nb = 0; nb < 8; ++nb) {
        uintx8 bf = load_b_frag(WtU, 128 * wv + 16 * nb, k0, lane, W2T_STRIDE);
        Y[0][nb] = wmma_bf16(a0, bf, Y[0][nb]);
        Y[1][nb] = wmma_bf16(a1, bf, Y[1][nb]);
      }
    }
  }

  // --- epilogue: out[tok, col] += w_tok * (Y + b2[col]) ---
#pragma unroll
  for (int m = 0; m < 2; ++m) {
#pragma unroll
    for (int nb = 0; nb < 8; ++nb) {
      int col = 128 * wv + 16 * nb + (lane & 15);
      float b2v = b2[e * D_DIM + col];
#pragma unroll
      for (int r = 0; r < 8; ++r) {
        int row = 16 * m + r + ((lane & 16) ? 8 : 0);
        int tok = toksL[row];
        float wt = wtsL[row];
        float val = wt * (Y[m][nb][r] + b2v);
        atomicAdd(&out[(size_t)tok * D_DIM + col], val);
      }
    }
  }
}

// ---------------------------------------------------------------------------
// Host launcher
// ---------------------------------------------------------------------------
extern "C" void kernel_launch(void* const* d_in, const int* in_sizes, int n_in,
                              void* d_out, int out_size, void* d_ws, size_t ws_size,
                              hipStream_t stream) {
  const float* x  = (const float*)d_in[0];   // [S,B,D]
  const float* gw = (const float*)d_in[1];   // [E,D]
  const float* gb = (const float*)d_in[2];   // [E]
  const float* w1 = (const float*)d_in[3];   // [E,D,F]
  const float* b1 = (const float*)d_in[4];   // [E,F]
  const float* w2 = (const float*)d_in[5];   // [E,F,D]
  const float* b2 = (const float*)d_in[6];   // [E,D]
  float* out = (float*)d_out;

  const int n_x  = TOKENS * D_DIM;             // 16,777,216
  const int n_w  = N_EXPERTS * D_DIM * F_DIM;  // 33,554,432 (per weight tensor)

  // workspace layout (all offsets 16B+ aligned): ~161 MB total
  char* ws = (char*)d_ws;
  size_t off = 0;
  u16* xb  = (u16*)(ws + off); off += (size_t)n_x * 2;   // 32 MB
  u16* w1t = (u16*)(ws + off); off += (size_t)n_w * 2;   // 64 MB  [e][f][k]
  u16* w2t = (u16*)(ws + off); off += (size_t)n_w * 2;   // 64 MB  [e][d][f]
  int* cnt   = (int*)(ws + off); off += 256;
  int* lists = (int*)(ws + off); off += (size_t)N_EXPERTS * TOKENS * 4;
  float* wlist = (float*)(ws + off); off += (size_t)N_EXPERTS * TOKENS * 4;
  (void)ws_size; (void)in_sizes; (void)n_in; (void)out_size;

  moe_zero_kernel<<<4096, 256, 0, stream>>>(out, n_x, cnt);
  moe_cvt_kernel<<<2048, 256, 0, stream>>>(x, xb, n_x);

  // w1 [e][k=1024][f=4096] -> w1t [e][f][k] ; w2 [e][f=4096][d=1024] -> w2t [e][d][f]
  dim3 trg1((D_DIM / 64) * (F_DIM / 64), N_EXPERTS);
  moe_cvt_tr_kernel<<<trg1, 256, 0, stream>>>(w1, w1t, D_DIM, F_DIM);
  dim3 trg2((F_DIM / 64) * (D_DIM / 64), N_EXPERTS);
  moe_cvt_tr_kernel<<<trg2, 256, 0, stream>>>(w2, w2t, F_DIM, D_DIM);

  moe_gate_kernel<<<TOKENS / 8, 256, 0, stream>>>(x, gw, gb, cnt, lists, wlist);

  dim3 grid(MAX_TILES, N_EXPERTS);
  moe_ffn_kernel<<<grid, 256, LDS_TOTAL, stream>>>(xb, w1t, w2t, b1, b2,
                                                   cnt, lists, wlist, out);
}
